// HGNN_conv_79663053406709
// MI455X (gfx1250) — compile-verified, matrix operand order checked by reading, not run on hardware
//
#include <hip/hip_runtime.h>
#include <hip/hip_bf16.h>

// ---------------- problem constants (fixed by the reference) ----------------
#define T_DIM   8
#define N_DIM   1024
#define E0_DIM  256
#define E1_DIM  256
#define E_DIM   512      // E0 + E1
#define F_DIM   64       // IN_FT == OUT_FT == 64
#define REP     28
#define BS      (REP * T_DIM)   // 224
#define EPS     1e-6f

// ---------------- WMMA vector types (probe-confirmed signatures) ------------
typedef __bf16 v16bf __attribute__((ext_vector_type(16)));
typedef __bf16 v8bf  __attribute__((ext_vector_type(8)));
typedef float  v8f   __attribute__((ext_vector_type(8)));

union FragBF { v16bf v; v8bf h[2]; };

// Load a 16x32 (MxK) bf16 A/B fragment for v_wmma_f32_16x16x32_bf16 from a
// row-major [row][K] matrix.  Per ISA 7.12.2 (16-bit A 16x32):
//   lanes 0-15  hold M=lane,    K = {0..7} (VGPR0-3) and {16..23} (VGPR4-7)
//   lanes 16-31 hold M=lane-16, K = {8..15}          and {24..31}
// rowBase must already point at this lane's row (row = lane & 15).
__device__ __forceinline__ v16bf load_frag_row(const __hip_bfloat16* rowBase,
                                               int k0, int lane) {
  const int half = (lane >> 4) << 3;   // 0 or 8
  FragBF f;
  f.h[0] = *(const v8bf*)(rowBase + k0 + half);        // global_load_b128
  f.h[1] = *(const v8bf*)(rowBase + k0 + 16 + half);   // global_load_b128
  return f.v;
}

#define WMMA_BF16(A, B, C) \
  __builtin_amdgcn_wmma_f32_16x16x32_bf16(false, (A), false, (B), (short)0, (C), false, false)

// Gc[t][n][e] = e < 256 ? G[n][e] : G1[t][n][e-256]
__device__ __forceinline__ float gc_at(const float* __restrict__ G,
                                       const float* __restrict__ G1,
                                       int t, int n, int e) {
  return (e < E0_DIM) ? G[n * E0_DIM + e]
                      : G1[((size_t)(t * N_DIM + n)) * E1_DIM + (e - E0_DIM)];
}

// ---------------- kernel 1: hyperedge degrees -> de_inv[T][E] ---------------
__global__ void de_kernel(const float* __restrict__ G,
                          const float* __restrict__ G1,
                          float* __restrict__ de_inv) {
  // grid = T * (E/256), block = 256; consecutive threads -> consecutive e
  const int t = blockIdx.x >> 1;
  const int e = ((blockIdx.x & 1) << 8) + threadIdx.x;
  float acc = 0.f;
  for (int n = 0; n < N_DIM; ++n) acc += gc_at(G, G1, t, n, e);
  de_inv[t * E_DIM + e] = 1.f / (acc + EPS);
}

// ---- kernel 2: vertex degrees + scaled bf16 operands Asc, Bsc [T][N][E] ----
//   Asc[n][e] = dv_is[n] * Gc[n][e] * de_inv[e]     (L = Asc @ Bsc^T)
//   Bsc[m][e] = Gc[m][e] * dv_is[m]
__global__ void prep_kernel(const float* __restrict__ G,
                            const float* __restrict__ G1,
                            const float* __restrict__ de_inv,
                            __hip_bfloat16* __restrict__ Asc,
                            __hip_bfloat16* __restrict__ Bsc) {
  // grid = T*N, block = E_DIM (=512)
  const int t = blockIdx.x >> 10;
  const int n = blockIdx.x & (N_DIM - 1);
  const int e = threadIdx.x;
  const float g = gc_at(G, G1, t, n, e);

  __shared__ float red[E_DIM];
  red[e] = g;
  __syncthreads();
  for (int s = E_DIM / 2; s > 0; s >>= 1) {
    if (e < s) red[e] += red[e + s];
    __syncthreads();
  }
  if (e == 0) red[0] = 1.f / sqrtf(red[0] + EPS);
  __syncthreads();
  const float dv_is = red[0];

  const size_t idx = ((size_t)(t * N_DIM + n)) * E_DIM + e;
  Asc[idx] = __float2bfloat16(g * de_inv[t * E_DIM + e] * dv_is);
  Bsc[idx] = __float2bfloat16(g * dv_is);
}

// ------- kernel 3: xw = x@W + b, stored TRANSPOSED bf16: xwT[s][f][n] -------
__global__ void xw_kernel(const float* __restrict__ x,
                          const float* __restrict__ W,
                          const float* __restrict__ bias,
                          __hip_bfloat16* __restrict__ xwT) {
  // grid = BS * (N/16), block = 256
  const int s     = blockIdx.x >> 6;          // batch sample 0..223
  const int ntile = blockIdx.x & 63;          // 16-row tile of N
  const int tid   = threadIdx.x;

  __shared__ float sW[F_DIM * F_DIM];         // 16 KB
  __shared__ float sX[16 * F_DIM];            // 4 KB
  for (int i = tid; i < F_DIM * F_DIM; i += 256) sW[i] = W[i];
  for (int i = tid; i < 16 * F_DIM; i += 256)
    sX[i] = x[((size_t)s * N_DIM + ntile * 16) * F_DIM + i];
  __syncthreads();

  for (int idx = tid; idx < 16 * F_DIM; idx += 256) {
    const int r = idx >> 6;                   // local row 0..15
    const int f = idx & (F_DIM - 1);          // output feature
    float acc = bias[f];
    #pragma unroll 8
    for (int k = 0; k < F_DIM; ++k) acc += sX[r * F_DIM + k] * sW[k * F_DIM + f];
    xwT[((size_t)s * F_DIM + f) * N_DIM + ntile * 16 + r] = __float2bfloat16(acc);
  }
}

// ---- kernel 4: L[t] = Asc[t] (1024x512) x Bsc[t]^T -> bf16 [T][N][N] -------
// 32x32 tile per wave (2x2 register blocking): 16 flops/byte from L2.
__global__ void gemm1_kernel(const __hip_bfloat16* __restrict__ Asc,
                             const __hip_bfloat16* __restrict__ Bsc,
                             __hip_bfloat16* __restrict__ Lbf) {
  // wave-tiles = T * 32 * 32 = 8192 ; 4 waves / block
  const int lane = threadIdx.x & 31;
  const int gt   = blockIdx.x * 4 + (threadIdx.x >> 5);
  const int t    = gt >> 10;                  // / 1024
  const int rem  = gt & 1023;
  const int tm   = rem >> 5;                  // 32-row block of M
  const int tn   = rem & 31;                  // 32-col block of N

  const __hip_bfloat16* a0 =
      Asc + ((size_t)(t * N_DIM + tm * 32 + (lane & 15))) * E_DIM;
  const __hip_bfloat16* a1 = a0 + (size_t)16 * E_DIM;
  const __hip_bfloat16* b0 =
      Bsc + ((size_t)(t * N_DIM + tn * 32 + (lane & 15))) * E_DIM;
  const __hip_bfloat16* b1 = b0 + (size_t)16 * E_DIM;

  v8f c00 = {}, c01 = {}, c10 = {}, c11 = {};
  for (int k0 = 0; k0 < E_DIM; k0 += 32) {
    __builtin_prefetch(a0 + k0 + 64, 0, 1);   // global_prefetch_b8
    __builtin_prefetch(b0 + k0 + 64, 0, 1);
    const v16bf va0 = load_frag_row(a0, k0, lane);
    const v16bf va1 = load_frag_row(a1, k0, lane);
    const v16bf vb0 = load_frag_row(b0, k0, lane);
    const v16bf vb1 = load_frag_row(b1, k0, lane);
    c00 = WMMA_BF16(va0, vb0, c00);
    c01 = WMMA_BF16(va0, vb1, c01);
    c10 = WMMA_BF16(va1, vb0, c10);
    c11 = WMMA_BF16(va1, vb1, c11);
  }

  // D layout: VGPR r, lane l -> M = r + 8*(l>=16), N = l&15
  __hip_bfloat16* lt = Lbf + (size_t)t * N_DIM * N_DIM;
  const int msub = ((lane >> 4) << 3);
  const int csub = (lane & 15);
  #pragma unroll
  for (int r = 0; r < 8; ++r) {
    const size_t m0 = (size_t)(tm * 32 + msub + r) * N_DIM;
    const size_t m1 = (size_t)(tm * 32 + 16 + msub + r) * N_DIM;
    lt[m0 + tn * 32 + csub]      = __float2bfloat16(c00[r]);
    lt[m0 + tn * 32 + 16 + csub] = __float2bfloat16(c01[r]);
    lt[m1 + tn * 32 + csub]      = __float2bfloat16(c10[r]);
    lt[m1 + tn * 32 + 16 + csub] = __float2bfloat16(c11[r]);
  }
}

// ---- kernel 5: out[s] (1024xF) = L[t(s)] (1024x1024) x xw[s] -> f32 --------
// 32n x 32f tile per wave (2x2 register blocking), K = 1024.
__global__ void gemm2_kernel(const __hip_bfloat16* __restrict__ Lbf,
                             const __hip_bfloat16* __restrict__ xwT,
                             float* __restrict__ out) {
  // wave-tiles = BS * (N/32) * (F/32) = 224*32*2 = 14336 ; 4 waves / block
  const int lane = threadIdx.x & 31;
  const int gt   = blockIdx.x * 4 + (threadIdx.x >> 5);
  const int s    = gt >> 6;                   // batch sample
  const int rem  = gt & 63;
  const int tn   = rem >> 1;                  // 32-row block of N
  const int tf   = rem & 1;                   // 32-col block of F
  const int t    = s / REP;                   // timestep

  const __hip_bfloat16* a0 =
      Lbf + (size_t)t * N_DIM * N_DIM +
      (size_t)(tn * 32 + (lane & 15)) * N_DIM;
  const __hip_bfloat16* a1 = a0 + (size_t)16 * N_DIM;
  const __hip_bfloat16* b0 =
      xwT + ((size_t)s * F_DIM + tf * 32 + (lane & 15)) * N_DIM;
  const __hip_bfloat16* b1 = b0 + (size_t)16 * N_DIM;

  v8f c00 = {}, c01 = {}, c10 = {}, c11 = {};
  for (int m0 = 0; m0 < N_DIM; m0 += 32) {
    __builtin_prefetch(a0 + m0 + 64, 0, 1);
    __builtin_prefetch(b0 + m0 + 64, 0, 1);
    const v16bf va0 = load_frag_row(a0, m0, lane);
    const v16bf va1 = load_frag_row(a1, m0, lane);
    const v16bf vb0 = load_frag_row(b0, m0, lane);
    const v16bf vb1 = load_frag_row(b1, m0, lane);
    c00 = WMMA_BF16(va0, vb0, c00);
    c01 = WMMA_BF16(va0, vb1, c01);
    c10 = WMMA_BF16(va1, vb0, c10);
    c11 = WMMA_BF16(va1, vb1, c11);
  }

  const int msub = ((lane >> 4) << 3);
  const int csub = (lane & 15);
  #pragma unroll
  for (int r = 0; r < 8; ++r) {
    const size_t n0 = ((size_t)s * N_DIM + tn * 32 + msub + r) * F_DIM;
    const size_t n1 = ((size_t)s * N_DIM + tn * 32 + 16 + msub + r) * F_DIM;
    out[n0 + tf * 32 + csub]      = c00[r];
    out[n0 + tf * 32 + 16 + csub] = c01[r];
    out[n1 + tf * 32 + csub]      = c10[r];
    out[n1 + tf * 32 + 16 + csub] = c11[r];
  }
}

// ------------------------------- launcher -----------------------------------
extern "C" void kernel_launch(void* const* d_in, const int* in_sizes, int n_in,
                              void* d_out, int out_size, void* d_ws, size_t ws_size,
                              hipStream_t stream) {
  const float* x    = (const float*)d_in[0];   // [224,1024,64]
  const float* G    = (const float*)d_in[1];   // [1024,256]
  const float* G1   = (const float*)d_in[2];   // [8,1024,256]
  const float* W    = (const float*)d_in[3];   // [64,64]
  const float* bias = (const float*)d_in[4];   // [64]
  float* out = (float*)d_out;                  // [224,1024,64] f32

  // workspace layout (all 16B-aligned, ~63 MB total)
  char* ws = (char*)d_ws;
  float*          de_inv = (float*)ws;                                   // 16 KB
  __hip_bfloat16* Asc    = (__hip_bfloat16*)(ws + (1 << 14));            // 8 MB
  __hip_bfloat16* Bsc    = Asc + (size_t)T_DIM * N_DIM * E_DIM;          // 8 MB
  __hip_bfloat16* Lbf    = Bsc + (size_t)T_DIM * N_DIM * E_DIM;          // 16 MB
  __hip_bfloat16* xwT    = Lbf + (size_t)T_DIM * N_DIM * N_DIM;          // 28.7 MB

  // 1) hyperedge degrees
  de_kernel<<<T_DIM * (E_DIM / 256), 256, 0, stream>>>(G, G1, de_inv);
  // 2) vertex degrees + scaled bf16 operands
  prep_kernel<<<T_DIM * N_DIM, E_DIM, 0, stream>>>(G, G1, de_inv, Asc, Bsc);
  // 3) xw = x@W + b (transposed bf16)
  xw_kernel<<<BS * (N_DIM / 16), 256, 0, stream>>>(x, W, bias, xwT);
  // 4) L = Asc @ Bsc^T  (bf16 WMMA 2x2-blocked, f32 accumulate, bf16 store)
  gemm1_kernel<<<(T_DIM * 32 * 32) / 4, 128, 0, stream>>>(Asc, Bsc, Lbf);
  // 5) out = L @ xw  (bf16 WMMA 2x2-blocked, f32 store)
  gemm2_kernel<<<(BS * 32 * 2) / 4, 128, 0, stream>>>(Lbf, xwT, out);
}
